// Contrast_2view_63110249448027
// MI455X (gfx1250) — matrix operand order repaired and checked by reading, not compiled, not run on hardware
//
#include <hip/hip_runtime.h>
#include <hip/hip_bf16.h>

#define TAU_F  0.5f
#define EPS_F  1e-8f
#define DDIM   256

typedef __attribute__((ext_vector_type(16))) __bf16 bf16x16;
typedef __attribute__((ext_vector_type(8)))  float  f32x8;

union Frag {
    bf16x16 v;
    uint4   q[2];
};

// ---------------- fp32 -> bf16 convert (x4 vectorized) ----------------
__global__ void cvt_f32_bf16(const float* __restrict__ src, __bf16* __restrict__ dst, int n4) {
    for (int i = blockIdx.x * blockDim.x + threadIdx.x; i < n4; i += gridDim.x * blockDim.x) {
        float4 f = ((const float4*)src)[i];
        union { __bf16 b[4]; uint2 u; } o;
        o.b[0] = (__bf16)f.x; o.b[1] = (__bf16)f.y;
        o.b[2] = (__bf16)f.z; o.b[3] = (__bf16)f.w;
        ((uint2*)dst)[i] = o.u;
    }
}

// ---------------- MLP layer: Out = act(A @ W^T + bias), all row-major ----------------
// A:[N,256] bf16, W:[256,256] bf16, bias:[256] f32, Out:[N,256] bf16
template<bool DO_ELU>
__global__ void __launch_bounds__(256, 1)
mlp_layer_kernel(const __bf16* __restrict__ A,
                 const __bf16* __restrict__ W,
                 const float*  __restrict__ bias,
                 __bf16* __restrict__ Out,
                 int N) {
    const int wave   = threadIdx.x >> 5;
    const int lane   = threadIdx.x & 31;
    const int hsel   = lane >> 4;      // which 16-lane half
    const int r      = lane & 15;
    const int istrip = blockIdx.x * 128 + wave * 16;
    const int arow   = istrip + r;

    // Preload the wave's 16x256 A strip as 8 WMMA fragments (registers).
    // bf16 A frag (16x32): lane(half h) holds K = {k0+8h..+7} then {k0+16+8h..+7}
    Frag af[8];
    const uint4* abase = (const uint4*)(A + (size_t)arow * DDIM);
#pragma unroll
    for (int kb = 0; kb < 8; ++kb) {
        const uint4* p = abase + kb * 4 + hsel;   // kb*32 bf16 = 4 uint4 ; hsel*8 bf16 = 1 uint4
        af[kb].q[0] = p[0];
        af[kb].q[1] = p[2];                       // +16 bf16 = +2 uint4
    }

#pragma unroll
    for (int nt = 0; nt < 16; ++nt) {
        const int n0  = nt * 16;
        const int col = n0 + r;                   // C/D: every lane owns one column
        const float bv = bias[col];
        f32x8 acc = { bv, bv, bv, bv, bv, bv, bv, bv };
        // bf16 B frag (32x16): lane(half h) = column 'col' of W^T = row 'col' of W,
        // K = {k0+8h..+7} then {k0+16+8h..+7}
        const uint4* wbase = (const uint4*)(W + (size_t)col * DDIM);
#pragma unroll
        for (int kb = 0; kb < 8; ++kb) {
            Frag bfr;
            const uint4* p = wbase + kb * 4 + hsel;
            bfr.q[0] = p[0];
            bfr.q[1] = p[2];
            acc = __builtin_amdgcn_wmma_f32_16x16x32_bf16(false, af[kb].v, false, bfr.v,
                                                          (short)0, acc, false, false);
        }
#pragma unroll
        for (int v = 0; v < 8; ++v) {
            float x = acc[v];
            if (DO_ELU) x = (x > 0.0f) ? x : (__expf(x) - 1.0f);   // ELU, alpha=1
            Out[(size_t)(istrip + v + 8 * hsel) * DDIM + col] = (__bf16)x;
        }
    }
}

// ---------------- reciprocal row L2 norms: rn[i] = 1 / ||P_i|| ----------------
__global__ void __launch_bounds__(256, 1)
rnorms_kernel(const __bf16* __restrict__ P, float* __restrict__ rn, int N) {
    const int wave = threadIdx.x >> 5;
    const int lane = threadIdx.x & 31;
    const int row  = blockIdx.x * 8 + wave;
    union { uint4 q; __bf16 b[8]; } u;
    u.q = ((const uint4*)(P + (size_t)row * DDIM))[lane];
    float s = 0.0f;
#pragma unroll
    for (int e = 0; e < 8; ++e) { float x = (float)u.b[e]; s += x * x; }
#pragma unroll
    for (int m = 16; m >= 1; m >>= 1) s += __shfl_xor(s, m, 32);
    if (lane == 0) rn[row] = rsqrtf(s);    // v_rsq_f32: reciprocal norm, no divides later
}

__global__ void zero_kernel(float* out) { out[0] = 0.0f; }

// ---------------- contrastive loss: dot GEMM + online exp-sum + diag ----------------
__global__ void __launch_bounds__(256, 1)
contrast_kernel(const __bf16* __restrict__ P1, const __bf16* __restrict__ P2,
                const float* __restrict__ rn1, const float* __restrict__ rn2,
                float* __restrict__ out, int N) {
    __shared__ __bf16 sB[128 * DDIM];             // 64 KB panel of P2 rows
    const int wave   = threadIdx.x >> 5;
    const int lane   = threadIdx.x & 31;
    const int hsel   = lane >> 4;
    const int r      = lane & 15;
    const int istrip = blockIdx.x * 128 + wave * 16;
    const int arow   = istrip + r;

    Frag af[8];
    const uint4* abase = (const uint4*)(P1 + (size_t)arow * DDIM);
#pragma unroll
    for (int kb = 0; kb < 8; ++kb) {
        const uint4* p = abase + kb * 4 + hsel;
        af[kb].q[0] = p[0];
        af[kb].q[1] = p[2];
    }

    float ascale[8];
#pragma unroll
    for (int v = 0; v < 8; ++v)
        ascale[v] = rn1[istrip + v + 8 * hsel] * (1.0f / TAU_F);

    float sume[8], diag[8];
#pragma unroll
    for (int v = 0; v < 8; ++v) { sume[v] = 0.0f; diag[v] = 0.0f; }

    for (int jp = 0; jp < N; jp += 128) {
        // stage P2[jp : jp+128, :] -> LDS (128*256 bf16 = 4096 uint4)
        {
            const uint4* gsrc = (const uint4*)(P2 + (size_t)jp * DDIM);
            uint4* sdst = (uint4*)sB;
            for (int t = threadIdx.x; t < 4096; t += 256) sdst[t] = gsrc[t];
        }
        if (jp + 128 < N)   // hint next panel toward the caches (global_prefetch_b8)
            __builtin_prefetch(P2 + (size_t)(jp + 128) * DDIM + threadIdx.x * 64, 0, 1);
        __syncthreads();

#pragma unroll
        for (int jt = 0; jt < 8; ++jt) {
            const int col = jp + jt * 16 + r;
            f32x8 acc = { 0.f, 0.f, 0.f, 0.f, 0.f, 0.f, 0.f, 0.f };
            const uint4* bbase = (const uint4*)(sB + (size_t)(jt * 16 + r) * DDIM);
#pragma unroll
            for (int kb = 0; kb < 8; ++kb) {
                Frag bfr;
                const uint4* p = bbase + kb * 4 + hsel;
                bfr.q[0] = p[0];
                bfr.q[1] = p[2];
                acc = __builtin_amdgcn_wmma_f32_16x16x32_bf16(false, af[kb].v, false, bfr.v,
                                                              (short)0, acc, false, false);
            }
            const float invn2 = rn2[col];          // plain load, no IEEE divide
#pragma unroll
            for (int v = 0; v < 8; ++v) {
                float sv = acc[v] * ascale[v] * invn2;
                sume[v] += __expf(sv);
                if (col == istrip + v + 8 * hsel) diag[v] = sv;   // diagonal element
            }
        }
        __syncthreads();
    }

    // reduce exp-sums / diag across the 16 column-lanes of each half
    float part = 0.0f;
#pragma unroll
    for (int v = 0; v < 8; ++v) {
        float se = sume[v], dg = diag[v];
#pragma unroll
        for (int m = 8; m >= 1; m >>= 1) {
            se += __shfl_xor(se, m, 32);
            dg += __shfl_xor(dg, m, 32);
        }
        if (r == 0) part += logf(se + EPS_F) - dg;   // -(log pos_i)
    }
    if (r == 0) atomicAdd(out, part / (float)N);
}

// ---------------- host launch ----------------
extern "C" void kernel_launch(void* const* d_in, const int* in_sizes, int n_in,
                              void* d_out, int out_size, void* d_ws, size_t ws_size,
                              hipStream_t stream) {
    const float* z1  = (const float*)d_in[0];
    const float* z2  = (const float*)d_in[1];
    const float* W1c = (const float*)d_in[2];
    const float* b1c = (const float*)d_in[3];
    const float* W2c = (const float*)d_in[4];
    const float* b2c = (const float*)d_in[5];
    const float* W1k = (const float*)d_in[6];
    const float* b1k = (const float*)d_in[7];
    const float* W2k = (const float*)d_in[8];
    const float* b2k = (const float*)d_in[9];
    const int N = in_sizes[0] / DDIM;          // 8192

    char* ws = (char*)d_ws;
    const size_t szP = (size_t)N * DDIM * sizeof(__bf16);       // 4 MB
    const size_t szW = (size_t)DDIM * DDIM * sizeof(__bf16);    // 128 KB
    __bf16* P1  = (__bf16*)(ws);
    __bf16* P2  = (__bf16*)(ws + szP);
    __bf16* H   = (__bf16*)(ws + 2 * szP);
    __bf16* ZB  = (__bf16*)(ws + 3 * szP);
    __bf16* W1B = (__bf16*)(ws + 4 * szP);
    __bf16* W2B = (__bf16*)(ws + 4 * szP + szW);
    float*  RN1 = (float*)(ws + 4 * szP + 2 * szW);
    float*  RN2 = (float*)(ws + 4 * szP + 2 * szW + (size_t)N * sizeof(float));

    dim3 b256(256);
    const int gemmBlocks = N / 128;

    // view 1: z1 -> P1
    cvt_f32_bf16<<<512, b256, 0, stream>>>(z1,  ZB,  N * DDIM / 4);
    cvt_f32_bf16<<<64,  b256, 0, stream>>>(W1c, W1B, DDIM * DDIM / 4);
    cvt_f32_bf16<<<64,  b256, 0, stream>>>(W2c, W2B, DDIM * DDIM / 4);
    mlp_layer_kernel<true ><<<gemmBlocks, b256, 0, stream>>>(ZB, W1B, b1c, H,  N);
    mlp_layer_kernel<false><<<gemmBlocks, b256, 0, stream>>>(H,  W2B, b2c, P1, N);

    // view 2: z2 -> P2
    cvt_f32_bf16<<<512, b256, 0, stream>>>(z2,  ZB,  N * DDIM / 4);
    cvt_f32_bf16<<<64,  b256, 0, stream>>>(W1k, W1B, DDIM * DDIM / 4);
    cvt_f32_bf16<<<64,  b256, 0, stream>>>(W2k, W2B, DDIM * DDIM / 4);
    mlp_layer_kernel<true ><<<gemmBlocks, b256, 0, stream>>>(ZB, W1B, b1k, H,  N);
    mlp_layer_kernel<false><<<gemmBlocks, b256, 0, stream>>>(H,  W2B, b2k, P2, N);

    rnorms_kernel<<<N / 8, b256, 0, stream>>>(P1, RN1, N);
    rnorms_kernel<<<N / 8, b256, 0, stream>>>(P2, RN2, N);
    zero_kernel<<<1, 1, 0, stream>>>((float*)d_out);
    contrast_kernel<<<gemmBlocks, b256, 0, stream>>>(P1, P2, RN1, RN2, (float*)d_out, N);
}
